// BondReactivityPredictor_23802708754731
// MI455X (gfx1250) — matrix operand order repaired and checked by reading, not compiled, not run on hardware
//
#include <hip/hip_runtime.h>
#include <stdint.h>

typedef __attribute__((ext_vector_type(16))) __bf16 v16bf;
typedef __attribute__((ext_vector_type(8)))  float  v8f;

#define MTILE 64
#define WAVES 8
#define LN_EPS 1e-5f

__device__ __forceinline__ uint32_t f2bf(float x) {
  uint32_t u = __float_as_uint(x);
  return (u + 0x7FFFu + ((u >> 16) & 1u)) >> 16;   // round-to-nearest-even
}
__device__ __forceinline__ uint32_t pack2(float lo, float hi) {
  return f2bf(lo) | (f2bf(hi) << 16);
}
__device__ __forceinline__ float fsigmoid(float x) { return 1.f / (1.f + __expf(-x)); }
__device__ __forceinline__ float fsilu(float x)    { return x * fsigmoid(x); }

// ---------------------------------------------------------------------------
// Prep kernels
// ---------------------------------------------------------------------------
__global__ void k_atom_sigmoid(const float* __restrict__ lg, float* __restrict__ out, int n) {
  int i = blockIdx.x * blockDim.x + threadIdx.x;
  if (i < n) out[i] = fsigmoid(lg[i]);
}

// node_embedding [N][128] f32 -> [N][64] u32 (bf16 pairs)
__global__ void k_node_pack(const float* __restrict__ ne, uint32_t* __restrict__ dst, int total) {
  int i = blockIdx.x * blockDim.x + threadIdx.x;
  if (i >= total) return;
  int n = i >> 6, j = i & 63;
  dst[i] = pack2(ne[n * 128 + 2 * j], ne[n * 128 + 2 * j + 1]);
}

// Swizzle f32 weight [K][rowLen] (rows rowOff..rowOff+Ksrc) into bf16 B-fragments.
// Fragment f = kt*nct + ct holds a 32x16 k-tile. Storage order within a fragment:
// u32 index = lane*8 + v, where VGPR v / lane follow the CDNA5 16-bit B layout:
//   lanes 0-15:  K = 2(v&3)+16(v>>2), +1  (N = lane)      lanes 16-31: K = 8+2(v&3)+16(v>>2), +1
__global__ void k_wswz(const float* __restrict__ src, int rowOff, int Ksrc, int rowLen,
                       int nct, uint32_t* __restrict__ dst, int total) {
  int idx = blockIdx.x * blockDim.x + threadIdx.x;
  if (idx >= total) return;
  int f = idx >> 8, rem = idx & 255;
  int lane = rem >> 3, v = rem & 7;
  int kt = f / nct, ct = f - kt * nct;
  int col = lane & 15, half = lane >> 4;
  int k0 = kt * 32 + (v & 3) * 2 + (v >> 2) * 16 + half * 8;
  int n = ct * 16 + col;
  float lo = (k0     < Ksrc) ? src[(size_t)(rowOff + k0)     * rowLen + n] : 0.f;
  float hi = (k0 + 1 < Ksrc) ? src[(size_t)(rowOff + k0 + 1) * rowLen + n] : 0.f;
  dst[idx] = pack2(lo, hi);
}

// ---------------------------------------------------------------------------
// WMMA GEMM over a 64-row tile: s_act[64][ldo] (+)= A(s_a bf16)[64][32*nkt] * W
// A in LDS (bf16 pairs, padded row stride ldaU u32); W fragments in global.
// Each wave owns PAIRS of column tiles (shared A fragment) and double-buffers
// the B fragments across the k-loop: two independent WMMAs per k-step.
// ---------------------------------------------------------------------------
__device__ __forceinline__ void gemm64(const uint32_t* __restrict__ sa, int ldaU,
                                       const uint4* __restrict__ wf, int nkt, int nct,
                                       float* __restrict__ sact, int ldo, bool accum) {
  const int lane = threadIdx.x & 31;
  const int wid  = threadIdx.x >> 5;
  const int col  = lane & 15;
  const int half = lane >> 4;
  const int hct  = nct >> 1;
  const int npairs = 4 * hct;
  const size_t kstride = (size_t)nct * 64;   // uint4 per k-step
  for (int t = wid; t < npairs; t += WAVES) {
    const int rt = t & 3, ct0 = t >> 2, ct1 = ct0 + hct;
    v8f c0, c1;
    if (accum) {
#pragma unroll
      for (int v = 0; v < 8; ++v) {
        c0[v] = sact[(rt * 16 + v + half * 8) * ldo + ct0 * 16 + col];
        c1[v] = sact[(rt * 16 + v + half * 8) * ldo + ct1 * 16 + col];
      }
    } else {
#pragma unroll
      for (int v = 0; v < 8; ++v) { c0[v] = 0.f; c1[v] = 0.f; }
    }
    // A layout (16-bit A 16x32): lane = M row (0-15 twice); half selects K+8
    const uint32_t* abase = sa + (rt * 16 + col) * ldaU + half * 4;
    const uint4* bp0 = wf + (size_t)ct0 * 64 + lane * 2;
    const uint4* bp1 = wf + (size_t)ct1 * 64 + lane * 2;
    // prime double buffer with k-step 0
    uint4 b00 = bp0[0], b01 = bp0[1];
    uint4 b10 = bp1[0], b11 = bp1[1];
    for (int kt = 0; kt < nkt; ++kt) {
      union { uint32_t u[8]; v16bf v; } A, B0, B1;
#pragma unroll
      for (int v = 0; v < 8; ++v)
        A.u[v] = abase[kt * 16 + ((v & 4) << 1) + (v & 3)];
      B0.u[0] = b00.x; B0.u[1] = b00.y; B0.u[2] = b00.z; B0.u[3] = b00.w;
      B0.u[4] = b01.x; B0.u[5] = b01.y; B0.u[6] = b01.z; B0.u[7] = b01.w;
      B1.u[0] = b10.x; B1.u[1] = b10.y; B1.u[2] = b10.z; B1.u[3] = b10.w;
      B1.u[4] = b11.x; B1.u[5] = b11.y; B1.u[6] = b11.z; B1.u[7] = b11.w;
      if (kt + 1 < nkt) {           // prefetch next k-step into registers
        const uint4* n0 = bp0 + (size_t)(kt + 1) * kstride;
        const uint4* n1 = bp1 + (size_t)(kt + 1) * kstride;
        b00 = n0[0]; b01 = n0[1];
        b10 = n1[0]; b11 = n1[1];
      }
      c0 = __builtin_amdgcn_wmma_f32_16x16x32_bf16(false, A.v, false, B0.v,
                                                   (short)0, c0, false, false);
      c1 = __builtin_amdgcn_wmma_f32_16x16x32_bf16(false, A.v, false, B1.v,
                                                   (short)0, c1, false, false);
    }
#pragma unroll
    for (int v = 0; v < 8; ++v) {
      sact[(rt * 16 + v + half * 8) * ldo + ct0 * 16 + col] = c0[v];
      sact[(rt * 16 + v + half * 8) * ldo + ct1 * 16 + col] = c1[v];
    }
  }
}

// bias + LayerNorm + SiLU over 128-wide rows; write bf16 A operand
__device__ __forceinline__ void ln_silu_ep(const float* __restrict__ sact,
                                           const float* __restrict__ bias,
                                           const float* __restrict__ g,
                                           const float* __restrict__ bb,
                                           uint32_t* __restrict__ dstA, int ldaU,
                                           float2* __restrict__ s_red) {
  const int t = threadIdx.x;
  const int r = t >> 2, sub = t & 3;
  const float* row = sact + r * 128;
  const int c0 = sub * 32;
  float s = 0.f, s2 = 0.f;
#pragma unroll
  for (int i = 0; i < 32; ++i) {
    float x = row[c0 + i] + bias[c0 + i];
    s += x; s2 += x * x;
  }
  s_red[t] = make_float2(s, s2);
  __syncthreads();
  float S = 0.f, S2 = 0.f;
#pragma unroll
  for (int i = 0; i < 4; ++i) { float2 p = s_red[(r << 2) | i]; S += p.x; S2 += p.y; }
  const float mean = S * (1.f / 128.f);
  const float var  = S2 * (1.f / 128.f) - mean * mean;
  const float rstd = rsqrtf(var + LN_EPS);
#pragma unroll
  for (int j = 0; j < 16; ++j) {
    int c = c0 + 2 * j;
    float x0 = row[c]     + bias[c];
    float x1 = row[c + 1] + bias[c + 1];
    float y0 = (x0 - mean) * rstd * g[c]     + bb[c];
    float y1 = (x1 - mean) * rstd * g[c + 1] + bb[c + 1];
    dstA[r * ldaU + (c0 >> 1) + j] = pack2(fsilu(y0), fsilu(y1));
  }
}

// LDS layout (bytes)
#define O_ACT   0        // 64*256 f32          = 65536
#define O_A     65536    // 64*129 u32          = 33024
#define O_E     98560    // 64*65  u32          = 16640
#define O_SE    115200   // 64*65  u32          = 16640
#define O_DE    131840   // 64*65  u32          = 16640
#define O_RED   148480   // 256 float2          = 2048
#define O_SRC   150528
#define O_DST   150784
#define O_PS    151040
#define O_PD    151296
#define O_DP    151552
#define SMEM_BYTES 151808

__global__ __launch_bounds__(256)
void fused_edge_kernel(const long long* __restrict__ eidx, int E,
                       const float* __restrict__ eattr,
                       const float* __restrict__ dualemb,
                       const uint32_t* __restrict__ nodebf,
                       const float* __restrict__ atomp,
                       const uint32_t* __restrict__ wbe1, const uint32_t* __restrict__ wbe2,
                       const uint32_t* __restrict__ wbe3,
                       const uint32_t* __restrict__ wd1, const uint32_t* __restrict__ wd2,
                       const uint32_t* __restrict__ wf1a, const uint32_t* __restrict__ wf1b,
                       const uint32_t* __restrict__ wf1c, const uint32_t* __restrict__ wf2,
                       const float* __restrict__ b_be1, const float* __restrict__ g_be1, const float* __restrict__ bb_be1,
                       const float* __restrict__ b_be2, const float* __restrict__ g_be2, const float* __restrict__ bb_be2,
                       const float* __restrict__ b_be3, const float* __restrict__ g_be3, const float* __restrict__ bb_be3,
                       const float* __restrict__ b_d1, const float* __restrict__ b_d2,
                       const float* __restrict__ W_do, const float* __restrict__ b_do,
                       const float* __restrict__ W_f1, const float* __restrict__ b_f1,
                       const float* __restrict__ b_f2,
                       const float* __restrict__ W_bo, const float* __restrict__ b_bo,
                       float* __restrict__ out) {
  extern __shared__ char smem[];
  float*    s_act = (float*)(smem + O_ACT);
  uint32_t* s_a   = (uint32_t*)(smem + O_A);
  uint32_t* s_e   = (uint32_t*)(smem + O_E);
  uint32_t* s_se  = (uint32_t*)(smem + O_SE);
  uint32_t* s_de  = (uint32_t*)(smem + O_DE);
  float2*   s_red = (float2*)(smem + O_RED);
  int*      s_src = (int*)(smem + O_SRC);
  int*      s_dst = (int*)(smem + O_DST);
  float*    s_ps  = (float*)(smem + O_PS);
  float*    s_pd  = (float*)(smem + O_PD);
  float*    s_dp  = (float*)(smem + O_DP);

  const int t = threadIdx.x;
  const long long e0 = (long long)blockIdx.x * MTILE;
  const int r = t >> 2, q = t & 3;
  const long long er = e0 + r < (long long)E ? e0 + r : (long long)E - 1;

  // --- edge indices + atom probabilities -----------------------------------
  if (t < MTILE) {
    long long ee = e0 + t < (long long)E ? e0 + t : (long long)E - 1;
    long long si = eidx[ee];
    long long di = eidx[(long long)E + ee];
    s_src[t] = (int)si; s_dst[t] = (int)di;
    s_ps[t] = atomp[si];
    s_pd[t] = atomp[di];
  }
  // --- edge_attr [64][16] -> s_a bf16 (K padded 16->32) --------------------
  {
    const float* ea = eattr + er * 16;
#pragma unroll
    for (int jj = 0; jj < 4; ++jj) {
      int j = q * 4 + jj;
      s_a[r * 17 + j] = (j < 8) ? pack2(ea[2 * j], ea[2 * j + 1]) : 0u;
    }
  }
  __syncthreads();
  // --- gather pre-converted bf16 node rows ---------------------------------
  {
    const uint32_t* srow = nodebf + (size_t)s_src[r] * 64 + q * 16;
    const uint32_t* drow = nodebf + (size_t)s_dst[r] * 64 + q * 16;
#pragma unroll
    for (int i = 0; i < 16; ++i) {
      s_se[r * 65 + q * 16 + i] = srow[i];
      s_de[r * 65 + q * 16 + i] = drow[i];
    }
  }
  // --- BondEmbedding --------------------------------------------------------
  gemm64(s_a, 17, (const uint4*)wbe1, 1, 8, s_act, 128, false);
  __syncthreads();
  ln_silu_ep(s_act, b_be1, g_be1, bb_be1, s_a, 65, s_red);
  __syncthreads();
  gemm64(s_a, 65, (const uint4*)wbe2, 4, 8, s_act, 128, false);
  __syncthreads();
  ln_silu_ep(s_act, b_be2, g_be2, bb_be2, s_a, 65, s_red);
  __syncthreads();
  gemm64(s_a, 65, (const uint4*)wbe3, 4, 8, s_act, 128, false);
  __syncthreads();
  ln_silu_ep(s_act, b_be3, g_be3, bb_be3, s_e, 65, s_red);   // stash e (bf16)
  // --- dual_node_emb [64][128] -> s_a bf16 (s_a is dead now) ---------------
  {
    const float* dn = dualemb + er * 128;
#pragma unroll
    for (int i = 0; i < 16; ++i) {
      int j = q * 16 + i;
      s_a[r * 65 + j] = pack2(dn[2 * j], dn[2 * j + 1]);
    }
  }
  __syncthreads();
  // --- dual predictor -------------------------------------------------------
  gemm64(s_a, 65, (const uint4*)wd1, 4, 16, s_act, 256, false);
  __syncthreads();
  {  // bias + SiLU -> s_a (K=256)
    const int c0 = q * 64;
    const float* row = s_act + r * 256;
#pragma unroll
    for (int j = 0; j < 32; ++j) {
      int c = c0 + 2 * j;
      s_a[r * 129 + (c0 >> 1) + j] =
          pack2(fsilu(row[c] + b_d1[c]), fsilu(row[c + 1] + b_d1[c + 1]));
    }
  }
  __syncthreads();
  gemm64(s_a, 129, (const uint4*)wd2, 8, 16, s_act, 256, false);
  __syncthreads();
  {  // bias + SiLU + dot with W_do
    const int c0 = q * 64;
    const float* row = s_act + r * 256;
    float acc = 0.f;
#pragma unroll
    for (int j = 0; j < 64; ++j) {
      int c = c0 + j;
      acc += fsilu(row[c] + b_d2[c]) * W_do[c];
    }
    s_red[t] = make_float2(acc, 0.f);
  }
  __syncthreads();
  if (q == 0)
    s_dp[r] = fsigmoid(s_red[t].x + s_red[t + 1].x + s_red[t + 2].x + s_red[t + 3].x + b_do[0]);
  __syncthreads();
  // --- main head layer 1: three accumulated GEMMs (concat split) -----------
  gemm64(s_se, 65, (const uint4*)wf1a, 4, 16, s_act, 256, false);
  gemm64(s_de, 65, (const uint4*)wf1b, 4, 16, s_act, 256, true);
  gemm64(s_e,  65, (const uint4*)wf1c, 4, 16, s_act, 256, true);
  __syncthreads();
  {  // + bias + rank-1 scalar features, SiLU -> s_a
    const float* row  = s_act + r * 256;
    const float* w384 = W_f1 + (size_t)384 * 256;
    const float* w385 = W_f1 + (size_t)385 * 256;
    const float* w386 = W_f1 + (size_t)386 * 256;
    const float dp = s_dp[r], ps = s_ps[r], pd = s_pd[r];
    const int c0 = q * 64;
#pragma unroll
    for (int j = 0; j < 32; ++j) {
      int c = c0 + 2 * j;
      float x0 = row[c]     + b_f1[c]     + dp * w384[c]     + ps * w385[c]     + pd * w386[c];
      float x1 = row[c + 1] + b_f1[c + 1] + dp * w384[c + 1] + ps * w385[c + 1] + pd * w386[c + 1];
      s_a[r * 129 + (c0 >> 1) + j] = pack2(fsilu(x0), fsilu(x1));
    }
  }
  __syncthreads();
  gemm64(s_a, 129, (const uint4*)wf2, 8, 16, s_act, 256, false);
  __syncthreads();
  {  // bias + SiLU + dot with W_bo
    const float* row = s_act + r * 256;
    const int c0 = q * 64;
    float acc = 0.f;
#pragma unroll
    for (int j = 0; j < 64; ++j) {
      int c = c0 + j;
      acc += fsilu(row[c] + b_f2[c]) * W_bo[c];
    }
    s_red[t] = make_float2(acc, 0.f);
  }
  __syncthreads();
  if (q == 0 && e0 + r < (long long)E)
    out[e0 + r] = s_red[t].x + s_red[t + 1].x + s_red[t + 2].x + s_red[t + 3].x + b_bo[0];
}

// ---------------------------------------------------------------------------
// Host launch
// ---------------------------------------------------------------------------
extern "C" void kernel_launch(void* const* d_in, const int* in_sizes, int n_in,
                              void* d_out, int out_size, void* d_ws, size_t ws_size,
                              hipStream_t stream) {
  const float*     node_emb = (const float*)d_in[0];
  const long long* eidx     = (const long long*)d_in[1];
  const float*     eattr    = (const float*)d_in[2];
  const float*     dualemb  = (const float*)d_in[3];
  const float*     logits   = (const float*)d_in[4];
  const float *W_be1 = (const float*)d_in[5],  *b_be1 = (const float*)d_in[6],
              *g_be1 = (const float*)d_in[7],  *bb_be1 = (const float*)d_in[8];
  const float *W_be2 = (const float*)d_in[9],  *b_be2 = (const float*)d_in[10],
              *g_be2 = (const float*)d_in[11], *bb_be2 = (const float*)d_in[12];
  const float *W_be3 = (const float*)d_in[13], *b_be3 = (const float*)d_in[14],
              *g_be3 = (const float*)d_in[15], *bb_be3 = (const float*)d_in[16];
  const float *W_d1 = (const float*)d_in[17], *b_d1 = (const float*)d_in[18];
  const float *W_d2 = (const float*)d_in[19], *b_d2 = (const float*)d_in[20];
  const float *W_do = (const float*)d_in[21], *b_do = (const float*)d_in[22];
  const float *W_f1 = (const float*)d_in[23], *b_f1 = (const float*)d_in[24];
  const float *W_f2 = (const float*)d_in[25], *b_f2 = (const float*)d_in[26];
  const float *W_bo = (const float*)d_in[27], *b_bo = (const float*)d_in[28];
  float* out = (float*)d_out;

  const int N = in_sizes[4];
  const int E = in_sizes[2] / 16;

  // workspace layout (u32 units)
  uint32_t* ws = (uint32_t*)d_ws;
  const size_t OFF_ATOMP = 0;                             // N f32 (pad 25088)
  const size_t OFF_NODE  = 25088;                         // N*64 u32
  const size_t OFF_WBE1  = OFF_NODE + (size_t)N * 64;     // 2048
  const size_t OFF_WBE2  = OFF_WBE1 + 2048;               // 8192
  const size_t OFF_WBE3  = OFF_WBE2 + 8192;               // 8192
  const size_t OFF_WD1   = OFF_WBE3 + 8192;               // 16384
  const size_t OFF_WD2   = OFF_WD1  + 16384;              // 32768
  const size_t OFF_WF1A  = OFF_WD2  + 32768;              // 16384
  const size_t OFF_WF1B  = OFF_WF1A + 16384;              // 16384
  const size_t OFF_WF1C  = OFF_WF1B + 16384;              // 16384
  const size_t OFF_WF2   = OFF_WF1C + 16384;              // 32768

  // prep: sigmoid(atom logits), node-emb -> bf16
  k_atom_sigmoid<<<(N + 255) / 256, 256, 0, stream>>>(logits, (float*)(ws + OFF_ATOMP), N);
  {
    int total = N * 64;
    k_node_pack<<<(total + 255) / 256, 256, 0, stream>>>(node_emb, ws + OFF_NODE, total);
  }
  // prep: swizzle weights into bf16 B-fragment layout
  struct Job { const float* src; int rowOff, Ksrc, rowLen, nct; size_t off; int total; };
  const Job jobs[10] = {
      {W_be1, 0,   16,  128, 8,  OFF_WBE1, 2048},
      {W_be2, 0,   128, 128, 8,  OFF_WBE2, 8192},
      {W_be3, 0,   128, 128, 8,  OFF_WBE3, 8192},
      {W_d1,  0,   128, 256, 16, OFF_WD1,  16384},
      {W_d2,  0,   256, 256, 16, OFF_WD2,  32768},
      {W_f1,  0,   128, 256, 16, OFF_WF1A, 16384},
      {W_f1,  128, 128, 256, 16, OFF_WF1B, 16384},
      {W_f1,  256, 128, 256, 16, OFF_WF1C, 16384},
      {W_f2,  0,   256, 256, 16, OFF_WF2,  32768},
      {nullptr, 0, 0, 0, 0, 0, 0}};
  for (int i = 0; i < 9; ++i) {
    const Job& j = jobs[i];
    k_wswz<<<(j.total + 255) / 256, 256, 0, stream>>>(j.src, j.rowOff, j.Ksrc, j.rowLen,
                                                      j.nct, ws + j.off, j.total);
  }
  // fused per-edge-tile kernel
  const int nblk = (E + MTILE - 1) / MTILE;
  fused_edge_kernel<<<nblk, 256, SMEM_BYTES, stream>>>(
      eidx, E, eattr, dualemb, ws + OFF_NODE, (const float*)(ws + OFF_ATOMP),
      ws + OFF_WBE1, ws + OFF_WBE2, ws + OFF_WBE3, ws + OFF_WD1, ws + OFF_WD2,
      ws + OFF_WF1A, ws + OFF_WF1B, ws + OFF_WF1C, ws + OFF_WF2,
      b_be1, g_be1, bb_be1, b_be2, g_be2, bb_be2, b_be3, g_be3, bb_be3,
      b_d1, b_d2, W_do, b_do, W_f1, b_f1, b_f2, W_bo, b_bo, out);
}
